// GRUD_55843164783425
// MI455X (gfx1250) — matrix operand order repaired and checked by reading, not compiled, not run on hardware
//
#include <hip/hip_runtime.h>
#include <hip/hip_bf16.h>

// ---------------- problem constants ----------------
#define B_    512
#define D_    256
#define T_    128
#define H_    1024
#define O_    64
#define HM_   341
#define GIN_  1536
#define BT_   16      // batch rows per workgroup (one WMMA M tile)
#define NTH_  512     // 16 waves of 32
#define SLOPE_ 0.01f

// output layout: [outputs B*O][tr_out B*T*O][h_final B*H]
#define TR_BASE_ ((size_t)B_ * O_)
#define HF_BASE_ (TR_BASE_ + (size_t)B_ * T_ * O_)

// ---------------- bf16 weight fragment workspace (element offsets) ----
// frag = 32 lanes x 16 bf16 = 512 elems; B-frag layout:
//   lane l, elem i  ->  K = kt*32 + (l/16)*8 + (i<8 ? i : i+8),  N = nt*16 + (l%16)
static constexpr size_t WZ_OFF  = 0;                     // 48*64*512
static constexpr size_t WR_OFF  = WZ_OFF  + (size_t)48*64*512;
static constexpr size_t WT_OFF  = WR_OFF  + (size_t)48*64*512;
static constexpr size_t WGH_OFF = WT_OFF  + (size_t)48*64*512;   // 8*64*512
static constexpr size_t WGX_OFF = WGH_OFF + (size_t)8*64*512;    // 8*16*512
static constexpr size_t WO1_OFF = WGX_OFF + (size_t)8*16*512;    // 32*22*512
static constexpr size_t WO2_OFF = WO1_OFF + (size_t)32*22*512;   // 11*4*512

typedef __attribute__((ext_vector_type(16))) __bf16 v16bf;
typedef __attribute__((ext_vector_type(8)))  float  v8f;

__device__ __forceinline__ v8f wmma_bf16(v16bf a, v16bf b, v8f c) {
  // D = A(16x32 bf16) x B(32x16 bf16) + C(16x16 f32)
  return __builtin_amdgcn_wmma_f32_16x16x32_bf16(false, a, false, b,
                                                 (short)0, c, false, false);
}

// scatter one value into A-fragment LDS layout:
// A frag: lane l (l%16 = M row; l/16 = K-group), elem i:
//   K_in_frag = (l/16)*8 + (i<8 ? i : i+8)
__device__ __forceinline__ void storeAfrag(__bf16* base, int K, int m, float v) {
  int f    = K >> 5;
  int kin  = K & 31;
  int lgrp = (kin >> 3) & 1;
  int i    = (kin & 7) | ((kin >> 4) << 3);
  base[(f * 32 + (m + 16 * lgrp)) * 16 + i] = (__bf16)v;
}

__device__ __forceinline__ float sigmoidf_(float x) {
  return 1.f / (1.f + __expf(-x));
}

// ---------------- weight pack: f32 row-major (K,N) -> bf16 B-fragments ----
__global__ void pack_w(const float* __restrict__ src, __bf16* __restrict__ dst,
                       int K, int N, int NT, int total) {
  for (int e = blockIdx.x * blockDim.x + threadIdx.x; e < total;
       e += gridDim.x * blockDim.x) {
    int frag = e >> 9;
    int r    = e & 511;
    int l    = r >> 4;
    int i    = r & 15;
    int kt   = frag / NT;
    int nt   = frag % NT;
    int sk   = kt * 32 + (l >> 4) * 8 + (i < 8 ? i : i + 8);
    int sn   = nt * 16 + (l & 15);
    float v  = (sk < K && sn < N) ? src[(size_t)sk * N + sn] : 0.f;
    dst[e] = (__bf16)v;
  }
}

// ---------------- persistent GRU-D kernel: one WG per 16 batch rows ------
__global__ __launch_bounds__(NTH_, 1) void grud_kernel(
    const float* __restrict__ inputs, const float* __restrict__ hidden,
    const float* __restrict__ x_mean,
    const float* __restrict__ b_z, const float* __restrict__ b_r,
    const float* __restrict__ b_t, const float* __restrict__ b_gx,
    const float* __restrict__ b_gh, const float* __restrict__ b_o1,
    const float* __restrict__ b_o2,
    const __bf16* __restrict__ wsb, float* __restrict__ out) {
  __shared__ float  hbuf[BT_ * H_];     // 64 KB, resident hidden state (f32)
  __shared__ __bf16 gfrag[48 * 512];    // 48 KB, gate_in A-fragments (K=1536)
  __shared__ __bf16 dfrag[8 * 512];     //  8 KB, delta A-fragments (K=256)
  __shared__ __bf16 y1frag[11 * 512];   // 11 KB, FC1 A-fragments (K=352)

  const int tid  = threadIdx.x;
  const int lane = tid & 31;
  const int w    = tid >> 5;           // wave id, owns H cols [64w, 64w+64)
  const int b0   = blockIdx.x * BT_;
  const int nlo  = lane & 15;
  const int mhi  = (lane >> 4) * 8;    // C-frag row offset

  const __bf16* WzF  = wsb + WZ_OFF;
  const __bf16* WrF  = wsb + WR_OFF;
  const __bf16* WtF  = wsb + WT_OFF;
  const __bf16* WghF = wsb + WGH_OFF;
  const __bf16* WgxF = wsb + WGX_OFF;
  const __bf16* Wo1F = wsb + WO1_OFF;
  const __bf16* Wo2F = wsb + WO2_OFF;

  // init hidden state
  for (int idx = tid; idx < BT_ * H_; idx += NTH_) {
    int r = idx >> 10;
    int c = idx & (H_ - 1);
    hbuf[idx] = hidden[(size_t)(b0 + r) * H_ + c];
  }
  __syncthreads();

#pragma unroll 1
  for (int t = 0; t < T_; ++t) {
    // ---- Phase A: load step inputs into fragments -----------------------
    // waves 0-7: delta -> dfrag[f];  waves 8-15: m -> gfrag K=1280+ section
    {
      int f    = w & 7;
      int c    = (w >= 8) ? 1 : 2;     // channel: m=1, delta=2
      int row  = nlo;
      int half = lane >> 4;
      size_t base =
          ((size_t)(b0 + row) * 4 + c) * ((size_t)T_ * D_) + (size_t)t * D_;
      alignas(32) __bf16 tmp[16];
#pragma unroll
      for (int i = 0; i < 16; ++i) {
        int kidx = half * 8 + (i < 8 ? i : i + 8);
        tmp[i] = (__bf16)inputs[base + f * 32 + kidx];
      }
      __bf16* dst = (w >= 8) ? (gfrag + (40 + f) * 512) : (dfrag + f * 512);
      *(v16bf*)(dst + lane * 16) = *(const v16bf*)tmp;
    }
    __syncthreads();

    // ---- Phase B: gamma_h, gamma_x matmuls; h *= gamma_h; build x_new ----
    {
      v8f gh[4] = {};
      v8f gx    = {};
#pragma unroll 1
      for (int f = 0; f < 8; ++f) {
        v16bf a = *(const v16bf*)(dfrag + (f * 32 + lane) * 16);
#pragma unroll
        for (int j = 0; j < 4; ++j) {
          v16bf bm = *(const v16bf*)(WghF +
                       ((size_t)(f * 64 + 4 * w + j) * 32 + lane) * 16);
          gh[j] = wmma_bf16(a, bm, gh[j]);
        }
        v16bf bx =
            *(const v16bf*)(WgxF + ((size_t)(f * 16 + w) * 32 + lane) * 16);
        gx = wmma_bf16(a, bx, gx);
      }
#pragma unroll
      for (int j = 0; j < 4; ++j) {
        int col  = (4 * w + j) * 16 + nlo;
        float bb = b_gh[col];
#pragma unroll
        for (int g = 0; g < 8; ++g) {
          int m     = g + mhi;
          float gam = __expf(-fmaxf(gh[j][g] + bb, 0.f));
          float hv  = hbuf[m * H_ + col] * gam;
          hbuf[m * H_ + col] = hv;
          storeAfrag(gfrag, 256 + col, m, hv);   // gate_in h-section
        }
      }
      {
        int col  = w * 16 + nlo;
        float bb = b_gx[col];
        float xm = x_mean[col];
#pragma unroll
        for (int g = 0; g < 8; ++g) {
          int m = g + mhi;
          size_t base =
              ((size_t)(b0 + m) * 4) * ((size_t)T_ * D_) + (size_t)t * D_;
          float xv  = inputs[base + col];                       // x     (c=0)
          float mv  = inputs[base + (size_t)T_ * D_ + col];     // m     (c=1)
          float x1  = inputs[base + 3 * (size_t)T_ * D_ + col]; // x_tm1 (c=3)
          float gxa = __expf(-fmaxf(gx[g] + bb, 0.f));
          float xn  = mv * xv + (1.f - mv) * (gxa * x1 + (1.f - gxa) * xm);
          storeAfrag(gfrag, col, m, xn);         // gate_in x-section
        }
      }
    }
    __syncthreads();

    // ---- Phase C: z, r gates (K=1536) -----------------------------------
    float zreg[4][8];
    {
      v8f az[4] = {};
      v8f ar[4] = {};
#pragma unroll 1
      for (int f = 0; f < 48; ++f) {
        v16bf a = *(const v16bf*)(gfrag + (f * 32 + lane) * 16);
        if (f + 1 < 48)
          __builtin_prefetch(
              WzF + ((size_t)((f + 1) * 64 + 4 * w) * 32 + lane) * 16, 0, 1);
#pragma unroll
        for (int j = 0; j < 4; ++j) {
          int nt = 4 * w + j;
          v16bf bz =
              *(const v16bf*)(WzF + ((size_t)(f * 64 + nt) * 32 + lane) * 16);
          az[j] = wmma_bf16(a, bz, az[j]);
          v16bf br =
              *(const v16bf*)(WrF + ((size_t)(f * 64 + nt) * 32 + lane) * 16);
          ar[j] = wmma_bf16(a, br, ar[j]);
        }
      }
      __syncthreads();  // all waves finished reading gfrag
#pragma unroll
      for (int j = 0; j < 4; ++j) {
        int col   = (4 * w + j) * 16 + nlo;
        float bbz = b_z[col];
        float bbr = b_r[col];
#pragma unroll
        for (int g = 0; g < 8; ++g) {
          int m = g + mhi;
          float zv   = sigmoidf_(az[j][g] + bbz);
          zreg[j][g] = zv;
          float rv   = sigmoidf_(ar[j][g] + bbr);
          storeAfrag(gfrag, 256 + col, m, rv * hbuf[m * H_ + col]); // r*h
        }
      }
    }
    __syncthreads();

    // ---- Phase D: tilde gate (K=1536) + h update -------------------------
    {
      v8f at4[4] = {};
#pragma unroll 1
      for (int f = 0; f < 48; ++f) {
        v16bf a = *(const v16bf*)(gfrag + (f * 32 + lane) * 16);
#pragma unroll
        for (int j = 0; j < 4; ++j) {
          v16bf bt_ = *(const v16bf*)(WtF +
                        ((size_t)(f * 64 + 4 * w + j) * 32 + lane) * 16);
          at4[j] = wmma_bf16(a, bt_, at4[j]);
        }
      }
      __syncthreads();  // matmul reads done before overwriting h-section
#pragma unroll
      for (int j = 0; j < 4; ++j) {
        int col  = (4 * w + j) * 16 + nlo;
        float bb = b_t[col];
#pragma unroll
        for (int g = 0; g < 8; ++g) {
          int m    = g + mhi;
          float tl = tanhf(at4[j][g] + bb);
          float zv = zreg[j][g];
          float hn = (1.f - zv) * hbuf[m * H_ + col] + zv * tl;
          hbuf[m * H_ + col] = hn;
          storeAfrag(gfrag, 256 + col, m, hn);  // bf16 h for FC (frags 8..39)
        }
      }
    }
    __syncthreads();

    // ---- Phase E: FC1  y1 = leakyrelu(h @ W_o1 + b_o1), N padded to 352 --
    {
      v8f a10 = {};
      v8f a11 = {};
#pragma unroll 1
      for (int f = 0; f < 32; ++f) {
        v16bf a = *(const v16bf*)(gfrag + ((8 + f) * 32 + lane) * 16);
        v16bf b1 =
            *(const v16bf*)(Wo1F + ((size_t)(f * 22 + w) * 32 + lane) * 16);
        a10 = wmma_bf16(a, b1, a10);
        if (w < 6) {
          v16bf b2 = *(const v16bf*)(Wo1F +
                       ((size_t)(f * 22 + 16 + w) * 32 + lane) * 16);
          a11 = wmma_bf16(a, b2, a11);
        }
      }
      {
        int col  = w * 16 + nlo;
        float bb = (col < HM_) ? b_o1[col] : 0.f;
#pragma unroll
        for (int g = 0; g < 8; ++g) {
          int m   = g + mhi;
          float y = a10[g] + bb;
          y = (y >= 0.f) ? y : SLOPE_ * y;
          storeAfrag(y1frag, col, m, y);
        }
      }
      if (w < 6) {
        int col  = (16 + w) * 16 + nlo;
        float bb = (col < HM_) ? b_o1[col] : 0.f;
#pragma unroll
        for (int g = 0; g < 8; ++g) {
          int m   = g + mhi;
          float y = a11[g] + bb;
          y = (y >= 0.f) ? y : SLOPE_ * y;
          storeAfrag(y1frag, col, m, y);
        }
      }
    }
    __syncthreads();

    // ---- Phase F: FC2  y2 = leakyrelu(y1 @ W_o2 + b_o2) -> tr_out --------
    if (w < 4) {
      v8f a2 = {};
#pragma unroll 1
      for (int f = 0; f < 11; ++f) {
        v16bf a = *(const v16bf*)(y1frag + (f * 32 + lane) * 16);
        v16bf bm =
            *(const v16bf*)(Wo2F + ((size_t)(f * 4 + w) * 32 + lane) * 16);
        a2 = wmma_bf16(a, bm, a2);
      }
      int col  = w * 16 + nlo;
      float bb = b_o2[col];
#pragma unroll
      for (int g = 0; g < 8; ++g) {
        int m   = g + mhi;
        float y = a2[g] + bb;
        y = (y >= 0.f) ? y : SLOPE_ * y;
        size_t b = (size_t)(b0 + m);
        out[TR_BASE_ + (b * T_ + t) * O_ + col] = y;
        if (t == T_ - 1) out[b * O_ + col] = y;  // outputs == fc(h_final)
      }
    }
    __syncthreads();
  }

  // h_final
  for (int idx = tid; idx < BT_ * H_; idx += NTH_) {
    int r = idx >> 10;
    int c = idx & (H_ - 1);
    out[HF_BASE_ + (size_t)(b0 + r) * H_ + c] = hbuf[idx];
  }
}

// ---------------- host launch ------------------------------------------
extern "C" void kernel_launch(void* const* d_in, const int* in_sizes, int n_in,
                              void* d_out, int out_size, void* d_ws,
                              size_t ws_size, hipStream_t stream) {
  (void)in_sizes; (void)n_in; (void)out_size; (void)ws_size;
  const float* inputs = (const float*)d_in[0];
  const float* hidden = (const float*)d_in[1];
  const float* x_mean = (const float*)d_in[2];
  const float* W_z  = (const float*)d_in[3];
  const float* b_z  = (const float*)d_in[4];
  const float* W_r  = (const float*)d_in[5];
  const float* b_r  = (const float*)d_in[6];
  const float* W_t  = (const float*)d_in[7];
  const float* b_t  = (const float*)d_in[8];
  const float* W_gx = (const float*)d_in[9];
  const float* b_gx = (const float*)d_in[10];
  const float* W_gh = (const float*)d_in[11];
  const float* b_gh = (const float*)d_in[12];
  const float* W_o1 = (const float*)d_in[13];
  const float* b_o1 = (const float*)d_in[14];
  const float* W_o2 = (const float*)d_in[15];
  const float* b_o2 = (const float*)d_in[16];
  __bf16* ws = (__bf16*)d_ws;
  float*  out = (float*)d_out;

  auto pack = [&](const float* src, size_t off, int K, int N, int KT, int NT) {
    int total = KT * NT * 512;
    pack_w<<<(total + 255) / 256, 256, 0, stream>>>(src, ws + off, K, N, NT,
                                                    total);
  };
  pack(W_z,  WZ_OFF,  GIN_, H_,  48, 64);
  pack(W_r,  WR_OFF,  GIN_, H_,  48, 64);
  pack(W_t,  WT_OFF,  GIN_, H_,  48, 64);
  pack(W_gh, WGH_OFF, D_,   H_,   8, 64);
  pack(W_gx, WGX_OFF, D_,   D_,   8, 16);
  pack(W_o1, WO1_OFF, H_,   HM_, 32, 22);
  pack(W_o2, WO2_OFF, HM_,  O_,  11,  4);

  grud_kernel<<<B_ / BT_, NTH_, 0, stream>>>(
      inputs, hidden, x_mean, b_z, b_r, b_t, b_gx, b_gh, b_o1, b_o2, ws, out);
}
